// PipelineMoEBlock_36086315221050
// MI455X (gfx1250) — matrix operand order repaired and checked by reading, not compiled, not run on hardware
//
#include <hip/hip_runtime.h>
#include <hip/hip_bf16.h>

typedef __attribute__((ext_vector_type(8)))  _Float16 v8h;
typedef __attribute__((ext_vector_type(16))) _Float16 v16h;
typedef __attribute__((ext_vector_type(8)))  float    v8f;
typedef __attribute__((ext_vector_type(4)))  int      v4i;

union HF16a { v16h v; v8h h[2]; };

// ---- gfx1250 async global->LDS copy (ASYNCcnt-tracked), guarded fallback ----
#if __has_builtin(__builtin_amdgcn_global_load_async_to_lds_b128)
#define HAVE_ASYNC_LDS 1
typedef __attribute__((address_space(1))) v4i v4i_g;  // global (__device__) int4
typedef __attribute__((address_space(3))) v4i v4i_l;  // LDS (__shared__) int4
__device__ __forceinline__ void async_copy16(const _Float16* g, _Float16* l) {
    __builtin_amdgcn_global_load_async_to_lds_b128((v4i_g*)g, (v4i_l*)l, 0, 0);
}
__device__ __forceinline__ void async_wait0() {
#if __has_builtin(__builtin_amdgcn_s_wait_asynccnt)
    __builtin_amdgcn_s_wait_asynccnt(0);
#else
    asm volatile("s_wait_asynccnt 0" ::: "memory");
#endif
}
#endif

// ---------------- elementwise f32 -> f16 convert ----------------
__global__ void cvt_f16_kernel(const float* __restrict__ in, _Float16* __restrict__ out, long n) {
    long i = ((long)blockIdx.x * blockDim.x + threadIdx.x) * 4;
    if (i + 3 < n) {
        out[i + 0] = (_Float16)in[i + 0];
        out[i + 1] = (_Float16)in[i + 1];
        out[i + 2] = (_Float16)in[i + 2];
        out[i + 3] = (_Float16)in[i + 3];
    }
}

// ---------------- LayerNorm over D=1024, f32 in -> f16 out ----------------
__global__ __launch_bounds__(256) void ln_f16_kernel(const float* __restrict__ x,
                                                     const float* __restrict__ sc,
                                                     const float* __restrict__ bi,
                                                     _Float16* __restrict__ out) {
    const int row = blockIdx.x;
    const int t = threadIdx.x;
    const float* xr = x + (size_t)row * 1024;
    float v[4], s = 0.f, ss = 0.f;
#pragma unroll
    for (int i = 0; i < 4; ++i) {
        v[i] = xr[t + i * 256];
        s += v[i];
        ss += v[i] * v[i];
    }
    __shared__ float r1[256], r2[256];
    r1[t] = s; r2[t] = ss;
    __syncthreads();
    for (int w = 128; w > 0; w >>= 1) {
        if (t < w) { r1[t] += r1[t + w]; r2[t] += r2[t + w]; }
        __syncthreads();
    }
    const float mu = r1[0] * (1.f / 1024.f);
    const float var = r2[0] * (1.f / 1024.f) - mu * mu;
    const float rstd = rsqrtf(var + 1e-5f);
    _Float16* orow = out + (size_t)row * 1024;
#pragma unroll
    for (int i = 0; i < 4; ++i) {
        const int c = t + i * 256;
        orow[c] = (_Float16)((v[i] - mu) * rstd * sc[c] + bi[c]);
    }
}

// ---------------- V transpose: v[b,s,h*64+d] -> vT[(b*16+h)*64+d][s] ----------------
__global__ void transp_v_kernel(const _Float16* __restrict__ v, _Float16* __restrict__ vT) {
    const long t = (long)blockIdx.x * 256 + threadIdx.x;  // over 8192*1024
    const int c = (int)(t & 1023);
    const long tok = t >> 10;
    const int s = (int)(tok & 1023);
    const int b = (int)(tok >> 10);
    const int h = c >> 6;
    const int d = c & 63;
    vT[((((size_t)(b * 16 + h)) * 64 + d) << 10) + s] = v[t];
}

// ---------------- Tiled WMMA GEMM: C[M,N] = A[M,K](f16) x W[N,K](f16)^T ----------------
// K-tile = 64, block tile 128x64, 8 waves each computing a 32x32 quadrant.
// EPI 0: outH = f*scale (f16)           EPI 1: outF = f + resid (f32)
// EPI 2: outH = gelu(f + bias[e*N+n])   EPI 3: outF = f + bias[e*N+n] + resid
template <int EPI>
__global__ __launch_bounds__(256) void gemm_wmma_kernel(
    const _Float16* __restrict__ A, const _Float16* __restrict__ Wt,
    const float* __restrict__ bias, const float* __restrict__ resid,
    float* __restrict__ outF, _Float16* __restrict__ outH,
    int N, int K, int chunkM, int eMask, float scale) {
    __shared__ __align__(16) _Float16 As[128 * 72];  // stride 72: 144B rows, 16B aligned
    __shared__ __align__(16) _Float16 Ws[64 * 72];

    const int t = threadIdx.x;
    const int lane = t & 31, wave = t >> 5;
    const int half = lane >> 4, lr = lane & 15;
    const int wy = wave >> 1, wx = wave & 1;
    const size_t rowBase = (size_t)blockIdx.z * chunkM + (size_t)blockIdx.y * 128;
    const int n0 = blockIdx.x * 64;
    const int e = (int)blockIdx.z & eMask;
    const _Float16* Wp = Wt + (size_t)e * N * K;

    v8f acc[2][2] = {};

    for (int kt = 0; kt < K; kt += 64) {
        __syncthreads();
        // cooperative stage: A tile 128x64 (4 chunks/thread), W tile 64x64 (2 chunks/thread)
        {
            int idx = t * 8;
#pragma unroll
            for (int p = 0; p < 4; ++p) {
                const int r = idx >> 6, c = idx & 63;
#ifdef HAVE_ASYNC_LDS
                async_copy16(&A[(rowBase + r) * (size_t)K + kt + c], &As[r * 72 + c]);
#else
                *(v8h*)&As[r * 72 + c] = *(const v8h*)&A[(rowBase + r) * (size_t)K + kt + c];
#endif
                idx += 2048;
            }
            idx = t * 8;
#pragma unroll
            for (int p = 0; p < 2; ++p) {
                const int r = idx >> 6, c = idx & 63;
#ifdef HAVE_ASYNC_LDS
                async_copy16(&Wp[(size_t)(n0 + r) * K + kt + c], &Ws[r * 72 + c]);
#else
                *(v8h*)&Ws[r * 72 + c] = *(const v8h*)&Wp[(size_t)(n0 + r) * K + kt + c];
#endif
                idx += 2048;
            }
        }
#ifdef HAVE_ASYNC_LDS
        async_wait0();  // our async writes landed in LDS
#endif
        __syncthreads();

#pragma unroll
        for (int kb = 0; kb < 2; ++kb) {
            HF16a fa[2], fb[2];
#pragma unroll
            for (int mi = 0; mi < 2; ++mi) {
                const _Float16* ap = &As[(wy * 32 + mi * 16 + lr) * 72 + kb * 32 + 8 * half];
                fa[mi].h[0] = *(const v8h*)ap;         // k = base .. base+7
                fa[mi].h[1] = *(const v8h*)(ap + 16);  // k = base+16 .. base+23
            }
#pragma unroll
            for (int ni = 0; ni < 2; ++ni) {
                const _Float16* bp = &Ws[(wx * 32 + ni * 16 + lr) * 72 + kb * 32 + 8 * half];
                fb[ni].h[0] = *(const v8h*)bp;         // k = base .. base+7
                fb[ni].h[1] = *(const v8h*)(bp + 16);  // k = base+16 .. base+23
            }
#pragma unroll
            for (int mi = 0; mi < 2; ++mi)
#pragma unroll
                for (int ni = 0; ni < 2; ++ni)
                    acc[mi][ni] = __builtin_amdgcn_wmma_f32_16x16x32_f16(
                        false, fa[mi].v, false, fb[ni].v, (short)0, acc[mi][ni], false, false);
        }
    }

    // epilogue (C layout: row = vgpr + 8*half, col = lane%16)
#pragma unroll
    for (int mi = 0; mi < 2; ++mi)
#pragma unroll
        for (int ni = 0; ni < 2; ++ni) {
            const int col = n0 + wx * 32 + ni * 16 + lr;
            const size_t r0 = rowBase + wy * 32 + mi * 16 + half * 8;
#pragma unroll
            for (int r = 0; r < 8; ++r) {
                const size_t idx = (r0 + r) * (size_t)N + col;
                const float f = acc[mi][ni][r];
                if constexpr (EPI == 0) {
                    outH[idx] = (_Float16)(f * scale);
                } else if constexpr (EPI == 1) {
                    outF[idx] = f + resid[idx];
                } else if constexpr (EPI == 2) {
                    const float u = f + bias[(size_t)e * N + col];
                    outH[idx] = (_Float16)(0.5f * u * (1.0f + erff(u * 0.70710678118f)));
                } else {
                    outF[idx] = f + bias[(size_t)e * N + col] + resid[idx];
                }
            }
        }
}

// ---------------- Flash attention: 1 wave per (b, h, 16-query strip) ----------------
// Row-sums of P are accumulated with an extra WMMA against a ones-matrix (same C
// layout as the O accumulator), leaving only the row-max as a cross-lane reduction.
__global__ __launch_bounds__(32) void attn_fa_kernel(const _Float16* __restrict__ q,
                                                     const _Float16* __restrict__ k,
                                                     const _Float16* __restrict__ vT,
                                                     _Float16* __restrict__ o) {
    __shared__ __align__(16) _Float16 Pt[16 * 40];
    const int lane = threadIdx.x & 31;
    const int half = lane >> 4, lr = lane & 15;
    const int qt = blockIdx.x, h = blockIdx.y, b = blockIdx.z;
    const int hoff = h * 64;
    const size_t qtok = (size_t)b * 1024 + qt * 16 + lr;

    HF16a qa[2];
#pragma unroll
    for (int kb = 0; kb < 2; ++kb) {
        const _Float16* qp = q + qtok * 1024 + hoff + kb * 32 + half * 8;
        qa[kb].h[0] = *(const v8h*)qp;
        qa[kb].h[1] = *(const v8h*)(qp + 16);
    }

    HF16a ones;
#pragma unroll
    for (int i = 0; i < 16; ++i) ones.v[i] = (_Float16)1.0f;

    float m[8];
#pragma unroll
    for (int r = 0; r < 8; ++r) m[r] = -1e30f;
    v8f accO[4] = {};
    v8f accS = {};  // row-sums of P (replicated across the 16 cols)

    for (int j = 0; j < 1024; j += 32) {
        v8f sc2[2];
#pragma unroll
        for (int jj = 0; jj < 2; ++jj) {
            const size_t ktok = (size_t)b * 1024 + j + jj * 16 + lr;
            v8f s = {};
#pragma unroll
            for (int kb = 0; kb < 2; ++kb) {
                HF16a kf;
                const _Float16* kp = k + ktok * 1024 + hoff + kb * 32 + half * 8;
                kf.h[0] = *(const v8h*)kp;
                kf.h[1] = *(const v8h*)(kp + 16);
                s = __builtin_amdgcn_wmma_f32_16x16x32_f16(false, qa[kb].v, false, kf.v,
                                                           (short)0, s, false, false);
            }
            sc2[jj] = s;
        }
        // row max across the 16 lanes of each half-wave (C-layout rows)
        float mt[8];
#pragma unroll
        for (int r = 0; r < 8; ++r) mt[r] = fmaxf(sc2[0][r], sc2[1][r]);
#pragma unroll
        for (int r = 0; r < 8; ++r)
#pragma unroll
            for (int d2 = 1; d2 < 16; d2 <<= 1) mt[r] = fmaxf(mt[r], __shfl_xor(mt[r], d2));
        float alpha[8];
#pragma unroll
        for (int r = 0; r < 8; ++r) {
            const float mn = fmaxf(m[r], mt[r]);
            alpha[r] = __expf(m[r] - mn);
            m[r] = mn;
        }
        float p0[8], p1[8];
#pragma unroll
        for (int r = 0; r < 8; ++r) {
            p0[r] = __expf(sc2[0][r] - m[r]);
            p1[r] = __expf(sc2[1][r] - m[r]);
        }
#pragma unroll
        for (int tt = 0; tt < 4; ++tt)
#pragma unroll
            for (int r = 0; r < 8; ++r) accO[tt][r] *= alpha[r];
#pragma unroll
        for (int r = 0; r < 8; ++r) accS[r] *= alpha[r];

        __syncthreads();  // single-wave WG: S_NOP, but orders LDS for compiler
#pragma unroll
        for (int r = 0; r < 8; ++r) {
            const int row = r + half * 8;
            Pt[row * 40 + lr] = (_Float16)p0[r];
            Pt[row * 40 + 16 + lr] = (_Float16)p1[r];
        }
        __syncthreads();
        HF16a pa;
        const _Float16* pp = &Pt[lr * 40 + half * 8];
        pa.h[0] = *(const v8h*)pp;
        pa.h[1] = *(const v8h*)(pp + 16);
        // l += rowsum(P) via WMMA against ones
        accS = __builtin_amdgcn_wmma_f32_16x16x32_f16(false, pa.v, false, ones.v,
                                                      (short)0, accS, false, false);
#pragma unroll
        for (int tt = 0; tt < 4; ++tt) {
            HF16a vf;
            const _Float16* vp =
                vT + ((size_t)(b * 16 + h) * 64 + tt * 16 + lr) * 1024 + j + half * 8;
            vf.h[0] = *(const v8h*)vp;
            vf.h[1] = *(const v8h*)(vp + 16);
            accO[tt] = __builtin_amdgcn_wmma_f32_16x16x32_f16(false, pa.v, false, vf.v,
                                                              (short)0, accO[tt], false, false);
        }
    }

#pragma unroll
    for (int tt = 0; tt < 4; ++tt)
#pragma unroll
        for (int r = 0; r < 8; ++r) {
            const size_t tok = (size_t)b * 1024 + qt * 16 + r + half * 8;
            o[tok * 1024 + hoff + tt * 16 + lr] = (_Float16)(accO[tt][r] / accS[r]);
        }
}

// ---------------- host orchestration ----------------
extern "C" void kernel_launch(void* const* d_in, const int* in_sizes, int n_in,
                              void* d_out, int out_size, void* d_ws, size_t ws_size,
                              hipStream_t stream) {
    (void)in_sizes; (void)n_in; (void)out_size; (void)ws_size;
    constexpr int NT = 8192, D = 1024, P = 4096, E = 8, S = 1024, Bb = 8, H = 16;

    const float* x       = (const float*)d_in[0];
    const float* aln_s   = (const float*)d_in[1];
    const float* aln_b   = (const float*)d_in[2];
    const float* wq      = (const float*)d_in[3];
    const float* wk      = (const float*)d_in[4];
    const float* wv      = (const float*)d_in[5];
    const float* wo      = (const float*)d_in[6];
    const float* mln_s   = (const float*)d_in[7];
    const float* mln_b   = (const float*)d_in[8];
    // d_in[9] = gate_w: top-k result is computed-then-discarded in the reference -> dead code
    const float* fc1w    = (const float*)d_in[10];
    const float* fc1b    = (const float*)d_in[11];
    const float* fc2w    = (const float*)d_in[12];
    const float* fc2b    = (const float*)d_in[13];
    float* out = (float*)d_out;

    char* ws = (char*)d_ws;
    size_t off = 0;
    auto take = [&](size_t bytes) -> char* {
        char* p = ws + off;
        off += (bytes + 255) & ~(size_t)255;
        return p;
    };
    _Float16* hA    = (_Float16*)take((size_t)NT * D * 2);
    _Float16* wq16  = (_Float16*)take((size_t)D * D * 2);
    _Float16* wk16  = (_Float16*)take((size_t)D * D * 2);
    _Float16* wv16  = (_Float16*)take((size_t)D * D * 2);
    _Float16* wo16  = (_Float16*)take((size_t)D * D * 2);
    _Float16* q16   = (_Float16*)take((size_t)NT * D * 2);
    _Float16* k16   = (_Float16*)take((size_t)NT * D * 2);
    _Float16* v16   = (_Float16*)take((size_t)NT * D * 2);
    _Float16* vT16  = (_Float16*)take((size_t)NT * D * 2);
    _Float16* o16   = (_Float16*)take((size_t)NT * D * 2);
    float*    xproc = (float*)   take((size_t)NT * D * 4);
    _Float16* xn16  = (_Float16*)take((size_t)NT * D * 2);
    _Float16* f1w16 = (_Float16*)take((size_t)E * P * D * 2);
    _Float16* f2w16 = (_Float16*)take((size_t)E * D * P * 2);
    _Float16* h1    = (_Float16*)take((size_t)NT * P * 2);

    // 1) weights -> f16
    cvt_f16_kernel<<<(D * D) / 1024, 256, 0, stream>>>(wq, wq16, (long)D * D);
    cvt_f16_kernel<<<(D * D) / 1024, 256, 0, stream>>>(wk, wk16, (long)D * D);
    cvt_f16_kernel<<<(D * D) / 1024, 256, 0, stream>>>(wv, wv16, (long)D * D);
    cvt_f16_kernel<<<(D * D) / 1024, 256, 0, stream>>>(wo, wo16, (long)D * D);
    cvt_f16_kernel<<<(E * P * D) / 1024, 256, 0, stream>>>(fc1w, f1w16, (long)E * P * D);
    cvt_f16_kernel<<<(E * D * P) / 1024, 256, 0, stream>>>(fc2w, f2w16, (long)E * D * P);

    // 2) LN1
    ln_f16_kernel<<<NT, 256, 0, stream>>>(x, aln_s, aln_b, hA);

    // 3) QKV projections (1/sqrt(64) folded into Q)
    dim3 gp(D / 64, NT / 128, 1);
    gemm_wmma_kernel<0><<<gp, 256, 0, stream>>>(hA, wq16, nullptr, nullptr, nullptr, q16,
                                                D, D, NT, 0, 0.125f);
    gemm_wmma_kernel<0><<<gp, 256, 0, stream>>>(hA, wk16, nullptr, nullptr, nullptr, k16,
                                                D, D, NT, 0, 1.0f);
    gemm_wmma_kernel<0><<<gp, 256, 0, stream>>>(hA, wv16, nullptr, nullptr, nullptr, v16,
                                                D, D, NT, 0, 1.0f);

    // 4) V transpose for contiguous PV B-fragments
    transp_v_kernel<<<(NT * D) / 256, 256, 0, stream>>>(v16, vT16);

    // 5) flash attention
    dim3 ga(S / 16, H, Bb);
    attn_fa_kernel<<<ga, 32, 0, stream>>>(q16, k16, vT16, o16);

    // 6) output projection + residual (f32)
    gemm_wmma_kernel<1><<<gp, 256, 0, stream>>>(o16, wo16, nullptr, x, xproc, nullptr,
                                                D, D, NT, 0, 1.0f);

    // 7) LN2
    ln_f16_kernel<<<NT, 256, 0, stream>>>(xproc, mln_s, mln_b, xn16);

    // 8) MoE fc1 + exact GELU (32 chunks of 256 tokens, expert = chunk & 7)
    dim3 g1(P / 64, 2, 32);
    gemm_wmma_kernel<2><<<g1, 256, 0, stream>>>(xn16, f1w16, fc1b, nullptr, nullptr, h1,
                                                P, D, 256, 7, 1.0f);

    // 9) MoE fc2 + bias + residual -> d_out
    dim3 g2(D / 64, 2, 32);
    gemm_wmma_kernel<3><<<g2, 256, 0, stream>>>(h1, f2w16, fc2b, xproc, out, nullptr,
                                                D, P, 256, 7, 1.0f);
}
